// SimplifiedDATBlock_64776696758421
// MI455X (gfx1250) — compile-verified, matrix operand order checked by reading, not run on hardware
//
#include <hip/hip_runtime.h>
#include <math.h>

// ---------------- types / helpers ----------------
typedef __attribute__((ext_vector_type(16))) __bf16 v16bf;
typedef __attribute__((ext_vector_type(8)))  float  v8f;
typedef __attribute__((ext_vector_type(4)))  unsigned int v4u;
typedef __attribute__((ext_vector_type(8)))  int    v8i;
typedef __attribute__((ext_vector_type(4)))  int    v4i;

__device__ __forceinline__ unsigned short f2bf(float f) {
    unsigned int u = __float_as_uint(f);
    u += 0x7FFFu + ((u >> 16) & 1u);          // round-to-nearest-even
    return (unsigned short)(u >> 16);
}
__device__ __forceinline__ float bf2f(unsigned short s) {
    return __uint_as_float(((unsigned int)s) << 16);
}
__device__ __forceinline__ __bf16 us_to_bf(unsigned short u) {
    union { unsigned short s; __bf16 b; } x; x.s = u; return x.b;
}
// A/B fragment K index for V_WMMA_*_16X16X32_BF16 (ISA 7.12.2, 16-bit A 16x32)
__device__ __forceinline__ int kidx(int e, int hi) {
    return ((e >> 3) << 4) + (((e >> 1) & 3) << 1) + (e & 1) + (hi << 3);
}
__device__ __forceinline__ float gelu_f(float x) {
    return 0.5f * x * (1.0f + erff(x * 0.7071067811865475f));
}

// ---------------- TDM availability ----------------
#if defined(__has_builtin)
#if __has_builtin(__builtin_amdgcn_tensor_load_to_lds) && __has_builtin(__builtin_amdgcn_s_wait_tensorcnt)
#define HAVE_TDM 1
#endif
#endif
#ifndef HAVE_TDM
#define HAVE_TDM 0
#endif

#if HAVE_TDM
// 2D bf16 tile load: rows x 32 elements, row stride = strideElems, zero-fill
// beyond (td0, td1). D# per CDNA5 ISA (group0: count/lds/global/type,
// group1: data_size, tensor_dim0/1, tile_dim0=32, tile_dim1=rows, dim0_stride).
__device__ __forceinline__ void tdm_load_2d(unsigned lds_byte, const void* gaddr,
                                            int td0, int td1, int rows, int strideElems) {
    unsigned long long ga = (unsigned long long)(uintptr_t)gaddr;
    v4u g0;
    g0[0] = 1u;                                            // count=1, user mode
    g0[1] = lds_byte;                                      // lds_addr
    g0[2] = (unsigned)ga;                                  // global_addr[31:0]
    g0[3] = (unsigned)((ga >> 32) & 0x1FFFFFFu) | (2u << 30); // addr[56:32] | type=2
    unsigned utd0 = (unsigned)(td0 < 0 ? 0 : td0);
    unsigned utd1 = (unsigned)(td1 < 0 ? 0 : td1);
    v8i g1;
    g1[0] = 1 << 16;                                       // data_size = 2 bytes
    g1[1] = (int)((utd0 & 0xFFFFu) << 16);                 // tensor_dim0[15:0] -> bits63:48
    g1[2] = (int)(((utd0 >> 16) & 0xFFFFu) | ((utd1 & 0xFFFFu) << 16));
    g1[3] = (int)(((utd1 >> 16) & 0xFFFFu) | (32u << 16)); // tile_dim0 = 32
    g1[4] = rows & 0xFFFF;                                 // tile_dim1 = rows, tile_dim2 = 0
    g1[5] = strideElems;                                   // tensor_dim0_stride[31:0]
    g1[6] = 0;
    g1[7] = 0;
    v4i z4 = {0, 0, 0, 0};
#if __clang_major__ >= 23
    v8i z8 = {0, 0, 0, 0, 0, 0, 0, 0};
    __builtin_amdgcn_tensor_load_to_lds(g0, g1, z4, z4, z8, 0);
#else
    __builtin_amdgcn_tensor_load_to_lds(g0, g1, z4, z4, 0);
#endif
}
#endif

// problem constants
#define BB    2
#define HH    256
#define WWID  256
#define LL    (HH * WWID)          // 65536
#define CC    180
#define NTOK  (BB * LL)            // 131072
#define NHEAD 6
#define HD    30
#define CQKV  (3 * CC)             // 540
#define FF    360
#define NWIN  (BB * (HH / 8) * (WWID / 8))  // 2048

// ---------------- LayerNorm (one token per block) ----------------
__global__ void ln_kernel(const float* __restrict__ in,
                          const float* __restrict__ g,
                          const float* __restrict__ bta,
                          float* __restrict__ outF,
                          unsigned short* __restrict__ outB) {
    __shared__ float red[256];
    const long base = (long)blockIdx.x * CC;
    const int  t    = threadIdx.x;
    float v = (t < CC) ? in[base + t] : 0.f;
    red[t] = v; __syncthreads();
    for (int s = 128; s > 0; s >>= 1) { if (t < s) red[t] += red[t + s]; __syncthreads(); }
    float mean = red[0] / (float)CC; __syncthreads();
    float d = (t < CC) ? (v - mean) : 0.f;
    red[t] = d * d; __syncthreads();
    for (int s = 128; s > 0; s >>= 1) { if (t < s) red[t] += red[t + s]; __syncthreads(); }
    float var = red[0] / (float)CC;
    float r = rsqrtf(var + 1e-5f);
    if (t < CC) {
        float o = d * r * g[t] + bta[t];
        if (outF) outF[base + t] = o;
        if (outB) outB[base + t] = f2bf(o);
    }
}

// ---------------- f32 -> bf16 weight conversion ----------------
__global__ void cvt_kernel(const float* __restrict__ in, unsigned short* __restrict__ out, int n) {
    int i = blockIdx.x * blockDim.x + threadIdx.x;
    if (i < n) out[i] = f2bf(in[i]);
}

// ---------------- bf16 WMMA GEMM:  out = [res +] act(A @ W^T + bias) ----------------
// A: (M,K) bf16 row-major. Wb: (N,K) bf16 row-major. Tile: 128(M) x 64(N), K-step 32.
// 4 waves; wave owns 2 M-tiles x 4 N-tiles (8 WMMAs / K-step). A/B tiles staged to
// LDS via TDM (tensor_load_to_lds) with hardware zero-fill at the K/N edges.
template <bool GELU>
__global__ void gemm_kernel(const unsigned short* __restrict__ A,
                            const unsigned short* __restrict__ Wb,
                            const float* __restrict__ bias,
                            const float* __restrict__ res,
                            float* __restrict__ outF,
                            unsigned short* __restrict__ outB,
                            int M, int N, int K) {
    __shared__ unsigned short As[128 * 32];
    __shared__ unsigned short Bs[64 * 32];
    const int m0   = blockIdx.x * 128;
    const int n0   = blockIdx.y * 64;
    const int tid  = threadIdx.x;
    const int lane = tid & 31;
    const int wave = tid >> 5;
    const int hi   = lane >> 4;
    const int ln   = lane & 15;

    v8f acc[2][4];
    #pragma unroll
    for (int mt = 0; mt < 2; mt++)
        #pragma unroll
        for (int nt = 0; nt < 4; nt++) acc[mt][nt] = v8f{0,0,0,0,0,0,0,0};

    for (int k0 = 0; k0 < K; k0 += 32) {
#if HAVE_TDM
        if (tid < 32) {   // wave 0 issues both DMA descriptors, waits, then all sync
            tdm_load_2d((unsigned)(uintptr_t)(void*)As, A  + (long)m0 * K + k0,
                        K - k0, M - m0, 128, K);
            tdm_load_2d((unsigned)(uintptr_t)(void*)Bs, Wb + (long)n0 * K + k0,
                        K - k0, N - n0, 64, K);
            __builtin_amdgcn_s_wait_tensorcnt(0);
        }
#else
        // fallback: dword-granular cooperative copy (K is even in all our calls)
        for (int d = tid; d < 128 * 16; d += 128) {
            int row = d >> 4, gk = k0 + ((d & 15) << 1);
            int gm = m0 + row;
            ((unsigned*)As)[d] = (gm < M && gk < K)
                ? *(const unsigned*)(A + (long)gm * K + gk) : 0u;
        }
        for (int d = tid; d < 64 * 16; d += 128) {
            int row = d >> 4, gk = k0 + ((d & 15) << 1);
            int gn = n0 + row;
            ((unsigned*)Bs)[d] = (gn < N && gk < K)
                ? *(const unsigned*)(Wb + (long)gn * K + gk) : 0u;
        }
#endif
        __syncthreads();
        v16bf af[2];
        #pragma unroll
        for (int mt = 0; mt < 2; mt++)
            #pragma unroll
            for (int e = 0; e < 16; e++)
                af[mt][e] = us_to_bf(As[(wave * 32 + mt * 16 + ln) * 32 + kidx(e, hi)]);
        #pragma unroll
        for (int nt = 0; nt < 4; nt++) {
            v16bf bfr;
            #pragma unroll
            for (int e = 0; e < 16; e++)
                bfr[e] = us_to_bf(Bs[(nt * 16 + ln) * 32 + kidx(e, hi)]);
            #pragma unroll
            for (int mt = 0; mt < 2; mt++)
                acc[mt][nt] = __builtin_amdgcn_wmma_f32_16x16x32_bf16(
                    false, af[mt], false, bfr, (short)0, acc[mt][nt], false, false);
        }
        __syncthreads();
    }
    #pragma unroll
    for (int nt = 0; nt < 4; nt++) {
        int n = n0 + nt * 16 + ln;
        if (n >= N) continue;
        float bv = bias ? bias[n] : 0.f;
        #pragma unroll
        for (int mt = 0; mt < 2; mt++) {
            #pragma unroll
            for (int r = 0; r < 8; r++) {
                int m = m0 + wave * 32 + mt * 16 + r + 8 * hi;
                if (m >= M) continue;
                float v = acc[mt][nt][r] + bv;
                if (GELU) v = gelu_f(v);
                if (res)  v += res[(long)m * N + n];
                if (outF) outF[(long)m * N + n] = v;
                if (outB) outB[(long)m * N + n] = f2bf(v);
            }
        }
    }
}

// ---------------- fused window attention: one (window, head) per block ----------------
__global__ void attn_kernel(const unsigned short* __restrict__ qkv,  // (NTOK, 540) bf16
                            const float* __restrict__ brel,          // (NH, 64, 64)
                            unsigned short* __restrict__ outb) {     // (NWIN*64, 180) bf16
    const int w  = blockIdx.x;            // window id
    const int h  = blockIdx.y;            // head id
    const int b  = w >> 10;
    const int wy = (w >> 5) & 31;
    const int wx = w & 31;
    const int tid  = threadIdx.x;
    const int lane = tid & 31;
    const int wave = tid >> 5;
    const int hi   = lane >> 4;
    const int ln   = lane & 15;

    __shared__ unsigned short qs[64 * 32], ks[64 * 32], vs[64 * 32];
    __shared__ float ps[64 * 66];
    __shared__ unsigned short pb[64 * 64];

    for (int idx = tid; idx < 64 * 32; idx += 128) {
        int t = idx >> 5, c = idx & 31;
        int iy = t >> 3, ix = t & 7;
        long l = (long)(wy * 8 + iy) * WWID + (wx * 8 + ix);
        const unsigned short* row = qkv + ((long)b * LL + l) * CQKV + h * HD;
        unsigned short qv = 0, kv = 0, vv = 0;
        if (c < HD) { qv = row[c]; kv = row[CC + c]; vv = row[2 * CC + c]; }
        qs[idx] = qv; ks[idx] = kv; vs[idx] = vv;
    }
    __syncthreads();

    {   // S = (Q * scale) K^T + brel
        v16bf af;
        #pragma unroll
        for (int e = 0; e < 16; e++) af[e] = us_to_bf(qs[(wave * 16 + ln) * 32 + kidx(e, hi)]);
        const float scale = 0.18257418583505536f;  // 30^-0.5
        #pragma unroll
        for (int nt = 0; nt < 4; nt++) {
            v16bf bfr;
            #pragma unroll
            for (int e = 0; e < 16; e++) bfr[e] = us_to_bf(ks[(nt * 16 + ln) * 32 + kidx(e, hi)]);
            v8f acc = v8f{0,0,0,0,0,0,0,0};
            acc = __builtin_amdgcn_wmma_f32_16x16x32_bf16(
                false, af, false, bfr, (short)0, acc, false, false);
            #pragma unroll
            for (int r = 0; r < 8; r++) {
                int mrow = wave * 16 + r + 8 * hi;
                int ncol = nt * 16 + ln;
                ps[mrow * 66 + ncol] = acc[r] * scale + brel[(long)h * 4096 + mrow * 64 + ncol];
            }
        }
    }
    __syncthreads();

    if (tid < 64) {  // per-row softmax
        float mx = -1e30f;
        for (int n = 0; n < 64; n++) mx = fmaxf(mx, ps[tid * 66 + n]);
        float s = 0.f;
        for (int n = 0; n < 64; n++) { float e = expf(ps[tid * 66 + n] - mx); ps[tid * 66 + n] = e; s += e; }
        float inv = 1.f / s;
        for (int n = 0; n < 64; n++) pb[tid * 64 + n] = f2bf(ps[tid * 66 + n] * inv);
    }
    __syncthreads();

    // O = P V   (K = 64, two K-steps; N = 30 padded to 32)
    v8f acc[2];
    acc[0] = v8f{0,0,0,0,0,0,0,0};
    acc[1] = v8f{0,0,0,0,0,0,0,0};
    for (int k0 = 0; k0 < 64; k0 += 32) {
        v16bf af;
        #pragma unroll
        for (int e = 0; e < 16; e++) af[e] = us_to_bf(pb[(wave * 16 + ln) * 64 + k0 + kidx(e, hi)]);
        #pragma unroll
        for (int nt = 0; nt < 2; nt++) {
            v16bf bfr;
            #pragma unroll
            for (int e = 0; e < 16; e++) bfr[e] = us_to_bf(vs[(k0 + kidx(e, hi)) * 32 + nt * 16 + ln]);
            acc[nt] = __builtin_amdgcn_wmma_f32_16x16x32_bf16(
                false, af, false, bfr, (short)0, acc[nt], false, false);
        }
    }
    #pragma unroll
    for (int nt = 0; nt < 2; nt++) {
        int c = nt * 16 + ln;
        if (c >= HD) continue;
        #pragma unroll
        for (int r = 0; r < 8; r++) {
            int mrow = wave * 16 + r + 8 * hi;
            outb[((long)w * 64 + mrow) * CC + h * HD + c] = f2bf(acc[nt][r]);
        }
    }
}

// ---------------- depthwise 3x3, channels-last ----------------
__global__ void dwconv_f32_gelu(const float* __restrict__ in, const float* __restrict__ kw,
                                float* __restrict__ out) {
    long idx = (long)blockIdx.x * blockDim.x + threadIdx.x;
    const long total = (long)NTOK * CC;
    if (idx >= total) return;
    int c = (int)(idx % CC);
    long t = idx / CC;
    int x = (int)(t & 255), y = (int)((t >> 8) & 255), b = (int)(t >> 16);
    float a = 0.f;
    #pragma unroll
    for (int dy = -1; dy <= 1; dy++) {
        int yy = y + dy; if ((unsigned)yy >= (unsigned)HH) continue;
        #pragma unroll
        for (int dx = -1; dx <= 1; dx++) {
            int xx = x + dx; if ((unsigned)xx >= (unsigned)WWID) continue;
            a += in[((long)b * LL + yy * WWID + xx) * CC + c] * kw[c * 9 + (dy + 1) * 3 + (dx + 1)];
        }
    }
    out[idx] = gelu_f(a);
}

__global__ void dwconv_bf16(const unsigned short* __restrict__ in, const float* __restrict__ kw,
                            unsigned short* __restrict__ out) {
    long idx = (long)blockIdx.x * blockDim.x + threadIdx.x;
    const long total = (long)NTOK * FF;
    if (idx >= total) return;
    int c = (int)(idx % FF);
    long t = idx / FF;
    int x = (int)(t & 255), y = (int)((t >> 8) & 255), b = (int)(t >> 16);
    float a = 0.f;
    #pragma unroll
    for (int dy = -1; dy <= 1; dy++) {
        int yy = y + dy; if ((unsigned)yy >= (unsigned)HH) continue;
        #pragma unroll
        for (int dx = -1; dx <= 1; dx++) {
            int xx = x + dx; if ((unsigned)xx >= (unsigned)WWID) continue;
            a += bf2f(in[((long)b * LL + yy * WWID + xx) * FF + c]) * kw[c * 9 + (dy + 1) * 3 + (dx + 1)];
        }
    }
    out[idx] = f2bf(a);
}

// ---------------- pooling (2-stage) + channel gate ----------------
__global__ void pool_partial(const float* __restrict__ convf, float* __restrict__ part) {
    int b = blockIdx.x >> 8, chunk = blockIdx.x & 255;
    int t = threadIdx.x;
    if (t >= CC) return;
    float s = 0.f;
    long base = ((long)b * LL + (long)chunk * 256) * CC + t;
    for (int i = 0; i < 256; i++) s += convf[base + (long)i * CC];
    part[(long)blockIdx.x * CC + t] = s;
}

__global__ void gate_kernel(const float* __restrict__ part,
                            const float* __restrict__ w1,   // (22,180)
                            const float* __restrict__ w2,   // (180,22)
                            float* __restrict__ cm) {       // (B,180)
    __shared__ float pooled[CC];
    __shared__ float t1[22];
    int b = blockIdx.x, t = threadIdx.x;
    if (t < CC) {
        float s = 0.f;
        for (int ch = 0; ch < 256; ch++) s += part[((long)(b << 8) + ch) * CC + t];
        pooled[t] = s / (float)LL;
    }
    __syncthreads();
    if (t < 22) {
        float s = 0.f;
        for (int c = 0; c < CC; c++) s += pooled[c] * w1[t * CC + c];
        t1[t] = gelu_f(s);
    }
    __syncthreads();
    if (t < CC) {
        float s = 0.f;
        for (int j = 0; j < 22; j++) s += t1[j] * w2[t * 22 + j];
        cm[b * CC + t] = 1.f / (1.f + expf(-s));
    }
}

// ---------------- AIM combine: x2 = x + attn_feat*cm + conv_feat ----------------
__global__ void combine_kernel(const float* __restrict__ x, const float* __restrict__ projf,
                               const float* __restrict__ convf, const float* __restrict__ cm,
                               float* __restrict__ x2) {
    long idx = (long)blockIdx.x * blockDim.x + threadIdx.x;
    const long total = (long)NTOK * CC;
    if (idx >= total) return;
    int c = (int)(idx % CC);
    long t = idx / CC;
    int xx = (int)(t & 255), yy = (int)((t >> 8) & 255), b = (int)(t >> 16);
    int wy = yy >> 3, iy = yy & 7, wx = xx >> 3, ix = xx & 7;
    long mrow = ((long)((b * 32 + wy) * 32 + wx)) * 64 + iy * 8 + ix;
    x2[idx] = x[idx] + projf[mrow * CC + c] * cm[b * CC + c] + convf[idx];
}

// ---------------- host launcher ----------------
static size_t al256(size_t x) { return (x + 255) & ~(size_t)255; }

extern "C" void kernel_launch(void* const* d_in, const int* in_sizes, int n_in,
                              void* d_out, int out_size, void* d_ws, size_t ws_size,
                              hipStream_t stream) {
    const float* x       = (const float*)d_in[0];
    const float* w_qkv   = (const float*)d_in[1];
    const float* b_rel   = (const float*)d_in[2];
    const float* w_proj  = (const float*)d_in[3];
    const float* b_proj  = (const float*)d_in[4];
    const float* conv_dw = (const float*)d_in[5];
    const float* cg_w1   = (const float*)d_in[6];
    const float* cg_w2   = (const float*)d_in[7];
    const float* ffn_w1  = (const float*)d_in[8];
    const float* ffn_dw  = (const float*)d_in[9];
    const float* ffn_w2  = (const float*)d_in[10];
    const float* n1_g    = (const float*)d_in[11];
    const float* n1_b    = (const float*)d_in[12];
    const float* n2_g    = (const float*)d_in[13];
    const float* n2_b    = (const float*)d_in[14];
    float* out = (float*)d_out;

    char* ws = (char*)d_ws;
    size_t off = 0;
    auto grab = [&](size_t bytes) { char* p = ws + off; off += al256(bytes); return p; };
    float*          n1f   = (float*)         grab((size_t)NTOK * CC * 4);
    unsigned short* n1b   = (unsigned short*)grab((size_t)NTOK * CC * 2);
    unsigned short* qkvb  = (unsigned short*)grab((size_t)NTOK * CQKV * 2);
    unsigned short* attnb = (unsigned short*)grab((size_t)NTOK * CC * 2);
    float*          projf = (float*)         grab((size_t)NTOK * CC * 4);
    float*          convf = (float*)         grab((size_t)NTOK * CC * 4);
    float*          poolp = (float*)         grab((size_t)BB * 256 * CC * 4);
    float*          cm    = (float*)         grab((size_t)BB * CC * 4);
    float*          x2    = (float*)         grab((size_t)NTOK * CC * 4);
    unsigned short* n2b   = (unsigned short*)grab((size_t)NTOK * CC * 2);
    unsigned short* hb    = (unsigned short*)grab((size_t)NTOK * FF * 2);
    unsigned short* hsb   = (unsigned short*)grab((size_t)NTOK * FF * 2);
    unsigned short* wqb   = (unsigned short*)grab((size_t)CQKV * CC * 2);
    unsigned short* wpb   = (unsigned short*)grab((size_t)CC * CC * 2);
    unsigned short* w1b   = (unsigned short*)grab((size_t)FF * CC * 2);
    unsigned short* w2b   = (unsigned short*)grab((size_t)CC * FF * 2);
    (void)ws_size; (void)in_sizes; (void)n_in; (void)out_size;

    // 0) one-time weight conversion f32 -> bf16 (tiny)
    cvt_kernel<<<(CQKV * CC + 255) / 256, 256, 0, stream>>>(w_qkv,  wqb, CQKV * CC);
    cvt_kernel<<<(CC * CC + 255) / 256,   256, 0, stream>>>(w_proj, wpb, CC * CC);
    cvt_kernel<<<(FF * CC + 255) / 256,   256, 0, stream>>>(ffn_w1, w1b, FF * CC);
    cvt_kernel<<<(CC * FF + 255) / 256,   256, 0, stream>>>(ffn_w2, w2b, CC * FF);

    // 1) LN1 -> n1 (f32 for conv branch, bf16 for QKV GEMM)
    ln_kernel<<<NTOK, 256, 0, stream>>>(x, n1_g, n1_b, n1f, n1b);
    // 2) QKV GEMM: (NTOK,180) @ (180,540)
    gemm_kernel<false><<<dim3(NTOK / 128, (CQKV + 63) / 64), 128, 0, stream>>>(
        n1b, wqb, nullptr, nullptr, nullptr, qkvb, NTOK, CQKV, CC);
    // 3) window attention (per window, per head) -> window-ordered bf16 rows
    attn_kernel<<<dim3(NWIN, NHEAD), 128, 0, stream>>>(qkvb, b_rel, attnb);
    // 4) proj GEMM (+bias) -> window-ordered f32
    gemm_kernel<false><<<dim3(NTOK / 128, (CC + 63) / 64), 128, 0, stream>>>(
        attnb, wpb, b_proj, nullptr, projf, nullptr, NTOK, CC, CC);
    // 5) conv branch: depthwise 3x3 + GELU on n1
    {
        long total = (long)NTOK * CC;
        dwconv_f32_gelu<<<(unsigned)((total + 255) / 256), 256, 0, stream>>>(n1f, conv_dw, convf);
    }
    // 6) pooled mean (2-stage) + channel gate
    pool_partial<<<BB * 256, 192, 0, stream>>>(convf, poolp);
    gate_kernel<<<BB, 256, 0, stream>>>(poolp, cg_w1, cg_w2, cm);
    // 7) x2 = x + attn_feat*cm + conv_feat
    {
        long total = (long)NTOK * CC;
        combine_kernel<<<(unsigned)((total + 255) / 256), 256, 0, stream>>>(x, projf, convf, cm, x2);
    }
    // 8) LN2 -> bf16
    ln_kernel<<<NTOK, 256, 0, stream>>>(x2, n2_g, n2_b, nullptr, n2b);
    // 9) FFN1 GEMM + GELU: (NTOK,180) @ (180,360)
    gemm_kernel<true><<<dim3(NTOK / 128, (FF + 63) / 64), 128, 0, stream>>>(
        n2b, w1b, nullptr, nullptr, nullptr, hb, NTOK, FF, CC);
    // 10) depthwise 3x3 on h (360 channels)
    {
        long total = (long)NTOK * FF;
        dwconv_bf16<<<(unsigned)((total + 255) / 256), 256, 0, stream>>>(hb, ffn_dw, hsb);
    }
    // 11) FFN2 GEMM + residual: out = x2 + h_s @ ffn_w2^T
    gemm_kernel<false><<<dim3(NTOK / 128, (CC + 63) / 64), 128, 0, stream>>>(
        hsb, w2b, nullptr, x2, out, nullptr, NTOK, CC, FF);
}
